// NeuralODE_16569983828134
// MI455X (gfx1250) — matrix operand order repaired and checked by reading, not compiled, not run on hardware
//
#include <hip/hip_runtime.h>
#include <hip/hip_bf16.h>

typedef _Float16 v16h __attribute__((ext_vector_type(16)));
typedef _Float16 v8h  __attribute__((ext_vector_type(8)));
typedef float    v8f  __attribute__((ext_vector_type(8)));

#define NPOS   65536      // 64 * 32 * 32
#define NELEM  524288     // 64 * 8 * 32 * 32

// ---------------- init: y0 = concat(x, zeros), scalars ----------------
__global__ __launch_bounds__(256) void k_init(float* __restrict__ y,
                                              const float* __restrict__ x,
                                              float* __restrict__ scal) {
  int idx = blockIdx.x * 256 + threadIdx.x;           // 0..524287 NCHW
  int b = idx >> 13, c = (idx >> 10) & 7, rest = idx & 1023;
  y[idx] = (c < 3) ? x[(b * 3 + c) * 1024 + rest] : 0.0f;
  if (idx == 0) { scal[0] = 0.0f; scal[1] = 0.1f; scal[2] = 0.0f; scal[3] = 0.0f; }
}

// ---------------- weight prep: f16 WMMA-A fragment layouts ----------------
// stored[m*32 + g*16 + e] holds K = 16*g + e   (contiguous k order, matches B)
__device__ __forceinline__ int frag_pos(int kl) {   // kl in 0..31 -> g*16+e
  return kl;
}

__global__ __launch_bounds__(256) void k_prep(const float* __restrict__ w1,
                                              const float* __restrict__ w2,
                                              const float* __restrict__ w3,
                                              _Float16* __restrict__ w1h,
                                              _Float16* __restrict__ w2h,
                                              _Float16* __restrict__ w3h,
                                              float* __restrict__ w1t,
                                              float* __restrict__ tw2,
                                              float* __restrict__ w3t) {
  int tid = threadIdx.x;
  for (int i = tid; i < 2048; i += 256) w1h[i] = (_Float16)0.0f;
  for (int i = tid; i < 1024; i += 256) w3h[i] = (_Float16)0.0f;
  __syncthreads();
  // conv1 weights: (64, 9, 1, 1); c=0 is time. K = c-1 (0..7), rest zero.
  for (int i = tid; i < 64 * 8; i += 256) {
    int o = i >> 3, K = i & 7;
    w1h[((o >> 4) * 16 + (o & 15)) * 32 + frag_pos(K)] = (_Float16)w1[o * 9 + K + 1];
  }
  for (int i = tid; i < 64; i += 256) w1t[i] = w1[i * 9];
  // conv2 weights: (64, 65, 3, 3); layout [tap][kc][mtile][512]
  for (int i = tid; i < 9 * 64 * 64; i += 256) {
    int tap = i / 4096, rem = i & 4095, o = rem >> 6, K = rem & 63;
    float v = w2[(o * 65 + K + 1) * 9 + tap];
    int kc = K >> 5, kl = K & 31;
    w2h[(((tap * 2 + kc) * 4 + (o >> 4)) << 9) + (o & 15) * 32 + frag_pos(kl)] = (_Float16)v;
  }
  // time-channel border-class sums for 3x3 conv (9 classes)
  for (int i = tid; i < 64 * 9; i += 256) {
    int o = i / 9, cls = i % 9, cy = cls / 3, cx = cls % 3;
    float s = 0.0f;
    for (int kh = 0; kh < 3; ++kh) {
      if ((cy == 0 && kh == 0) || (cy == 2 && kh == 2)) continue;
      for (int kw = 0; kw < 3; ++kw) {
        if ((cx == 0 && kw == 0) || (cx == 2 && kw == 2)) continue;
        s += w2[(o * 65) * 9 + kh * 3 + kw];
      }
    }
    tw2[o * 9 + cls] = s;
  }
  // conv3 weights: (8, 65, 1, 1); layout [kc][512], rows 8..15 zero
  for (int i = tid; i < 8 * 64; i += 256) {
    int o = i >> 6, K = i & 63;
    int kc = K >> 5, kl = K & 31;
    w3h[(kc << 9) + o * 32 + frag_pos(kl)] = (_Float16)w3[o * 65 + K + 1];
  }
  for (int i = tid; i < 8; i += 256) w3t[i] = w3[i * 65];
}

// ---------------- stage prep: ys = y + dtc*sum(a_j k_j) -> f16 NHWC(pad32) ----------------
__global__ __launch_bounds__(256) void k_stage(const float* __restrict__ y,
    const float* __restrict__ k1, const float* __restrict__ k2,
    const float* __restrict__ k3, const float* __restrict__ k4,
    const float* __restrict__ k5, const float* __restrict__ k6,
    float a1, float a2, float a3, float a4, float a5, float a6,
    _Float16* __restrict__ ys_h, float* __restrict__ y5out,
    const float* __restrict__ scal) {
  int tid = blockIdx.x * 256 + threadIdx.x;           // 0..2097151
  int p = tid >> 5, c = tid & 31;
  if (c >= 8) { ys_h[tid] = (_Float16)0.0f; return; }
  float t = scal[0], dt = scal[1];
  float dtc = fminf(dt, 1.0f - t);
  int idx = ((p >> 10) << 13) + (c << 10) + (p & 1023);  // NCHW flat
  float v = y[idx], s = 0.0f;
  if (a1 != 0.0f) s += a1 * k1[idx];
  if (a2 != 0.0f) s += a2 * k2[idx];
  if (a3 != 0.0f) s += a3 * k3[idx];
  if (a4 != 0.0f) s += a4 * k4[idx];
  if (a5 != 0.0f) s += a5 * k5[idx];
  if (a6 != 0.0f) s += a6 * k6[idx];
  v += dtc * s;
  ys_h[tid] = (_Float16)v;
  if (y5out) y5out[idx] = v;
}

// ---------------- conv1: 1x1, K=32 (8 live), register-blocked over 4 out-tiles ----------------
__global__ __launch_bounds__(256) void k_conv1(const _Float16* __restrict__ ys_h,
    const _Float16* __restrict__ w1h, const float* __restrict__ b1,
    const float* __restrict__ w1t, _Float16* __restrict__ h1_h,
    const float* __restrict__ scal, float ci) {
  int gw = (blockIdx.x * 256 + threadIdx.x) >> 5;     // 4096 pos tiles
  int lane = threadIdx.x & 31;
  int p0 = gw << 4, g = lane >> 4, n = lane & 15;
  v16h B = *(const v16h*)(ys_h + ((p0 + n) << 5) + (g << 4));
  v8f acc[4];
  const v8f z = {};
#pragma unroll
  for (int mt = 0; mt < 4; ++mt) {
    v16h A = *(const v16h*)(w1h + ((mt * 16 + n) << 5) + (g << 4));
    acc[mt] = __builtin_amdgcn_wmma_f32_16x16x32_f16(false, A, false, B, (short)0, z, false, false);
  }
  float t = scal[0], dt = scal[1];
  float ti = t + ci * fminf(dt, 1.0f - t);
  _Float16* outp = h1_h + ((p0 + n) << 6);
#pragma unroll
  for (int mt = 0; mt < 4; ++mt) {
    int chb = mt * 16 + g * 8;
    v8h outv;
#pragma unroll
    for (int r = 0; r < 8; ++r) {
      int ch = chb + r;
      float v = acc[mt][r] + b1[ch] + ti * w1t[ch];
      outv[r] = (_Float16)fmaxf(v, 0.0f);
    }
    *(v8h*)(outp + chb) = outv;
  }
}

// ---------------- conv2: 3x3 implicit GEMM, 9 taps x 2 K-chunks x 4 out-tiles ----------------
// One wave owns all 64 output channels for a 16-position tile: each B fragment
// load feeds 4 chained WMMAs (4x less activation traffic, 72 WMMAs/wave).
__global__ __launch_bounds__(256) void k_conv2(const _Float16* __restrict__ h1_h,
    const _Float16* __restrict__ w2h, const float* __restrict__ b2,
    const float* __restrict__ tw2, _Float16* __restrict__ h2_h,
    const float* __restrict__ scal, float ci) {
  int gw = (blockIdx.x * 256 + threadIdx.x) >> 5;     // 4096 pos tiles
  int lane = threadIdx.x & 31;
  int p0 = gw << 4;
  int b = p0 >> 10, yy = (p0 >> 5) & 31, x0 = p0 & 31;
  int g = lane >> 4, n = lane & 15;
  v8f acc[4] = {{}, {}, {}, {}};
  const v16h Bz = {};
#pragma unroll
  for (int tap = 0; tap < 9; ++tap) {
    int dy = tap / 3 - 1, dx = tap % 3 - 1;
    int yq = yy + dy;
    if (yq < 0 || yq > 31) continue;
    int xq = x0 + n + dx;
    bool ok = (xq >= 0) && (xq <= 31);
    int pq = (b * 32 + yq) * 32 + xq;
#pragma unroll
    for (int kc = 0; kc < 2; ++kc) {
      v16h B = ok ? *(const v16h*)(h1_h + (pq << 6) + (kc << 5) + (g << 4)) : Bz;
      const _Float16* wp = w2h + (((tap * 2 + kc) * 4) << 9) + (n << 5) + (g << 4);
#pragma unroll
      for (int mt = 0; mt < 4; ++mt) {
        v16h A = *(const v16h*)(wp + (mt << 9));
        acc[mt] = __builtin_amdgcn_wmma_f32_16x16x32_f16(false, A, false, B, (short)0, acc[mt], false, false);
      }
    }
  }
  float t = scal[0], dt = scal[1];
  float ti = t + ci * fminf(dt, 1.0f - t);
  int x = x0 + n;
  int cyc = (yy == 0) ? 0 : ((yy == 31) ? 2 : 1);
  int cxc = (x == 0) ? 0 : ((x == 31) ? 2 : 1);
  int cls = cyc * 3 + cxc;
  _Float16* outp = h2_h + ((p0 + n) << 6);
#pragma unroll
  for (int mt = 0; mt < 4; ++mt) {
    int chb = mt * 16 + g * 8;
    v8h outv;
#pragma unroll
    for (int r = 0; r < 8; ++r) {
      int ch = chb + r;
      float v = acc[mt][r] + b2[ch] + ti * tw2[ch * 9 + cls];
      outv[r] = (_Float16)fmaxf(v, 0.0f);
    }
    *(v8h*)(outp + chb) = outv;
  }
}

// ---------------- conv3: 1x1 -> k_i (f32 NCHW), 8 out channels ----------------
__global__ __launch_bounds__(256) void k_conv3(const _Float16* __restrict__ h2_h,
    const _Float16* __restrict__ w3h, const float* __restrict__ b3,
    const float* __restrict__ w3t, float* __restrict__ kout,
    const float* __restrict__ scal, float ci) {
  int gw = (blockIdx.x * 256 + threadIdx.x) >> 5;   // 4096 pos tiles
  int lane = threadIdx.x & 31;
  int p0 = gw << 4, g = lane >> 4, n = lane & 15;
  v8f acc = {};
#pragma unroll
  for (int kc = 0; kc < 2; ++kc) {
    v16h A = *(const v16h*)(w3h + (kc << 9) + (n << 5) + (g << 4));
    v16h B = *(const v16h*)(h2_h + ((p0 + n) << 6) + (kc << 5) + (g << 4));
    acc = __builtin_amdgcn_wmma_f32_16x16x32_f16(false, A, false, B, (short)0, acc, false, false);
  }
  if (g == 0) {
    float t = scal[0], dt = scal[1];
    float ti = t + ci * fminf(dt, 1.0f - t);
    int p = p0 + n;
    int base = ((p >> 10) << 13) + (p & 1023);
#pragma unroll
    for (int r = 0; r < 8; ++r)
      kout[base + (r << 10)] = acc[r] + b3[r] + ti * w3t[r];
  }
}

// ---------------- error estimate + deterministic two-phase reduction ----------------
__global__ __launch_bounds__(256) void k_err(const float* __restrict__ y,
    const float* __restrict__ y5,
    const float* __restrict__ k1, const float* __restrict__ k3,
    const float* __restrict__ k4, const float* __restrict__ k5,
    const float* __restrict__ k6, const float* __restrict__ k7,
    const float* __restrict__ scal, float* __restrict__ partial) {
  const float E1 = 71.0f/57600.0f, E3 = -71.0f/16695.0f, E4 = 71.0f/1920.0f,
              E5 = -17253.0f/339200.0f, E6 = 22.0f/525.0f, E7 = -1.0f/40.0f;
  __shared__ float sm[256];
  int idx = blockIdx.x * 256 + threadIdx.x;
  float t = scal[0], dt = scal[1];
  float dtc = fminf(dt, 1.0f - t);
  float e = dtc * (E1*k1[idx] + E3*k3[idx] + E4*k4[idx] + E5*k5[idx] + E6*k6[idx] + E7*k7[idx]);
  float tol = 1e-3f + 1e-3f * fmaxf(fabsf(y[idx]), fabsf(y5[idx]));
  float r = e / tol;
  sm[threadIdx.x] = r * r;
  __syncthreads();
  for (int s = 128; s > 0; s >>= 1) {
    if (threadIdx.x < s) sm[threadIdx.x] += sm[threadIdx.x + s];
    __syncthreads();
  }
  if (threadIdx.x == 0) partial[blockIdx.x] = sm[0];
}

__global__ __launch_bounds__(256) void k_control(float* __restrict__ scal,
                                                 const float* __restrict__ partial) {
  __shared__ float sm[256];
  float s = 0.0f;
  for (int i = threadIdx.x; i < 2048; i += 256) s += partial[i];
  sm[threadIdx.x] = s;
  __syncthreads();
  for (int st = 128; st > 0; st >>= 1) {
    if (threadIdx.x < st) sm[threadIdx.x] += sm[threadIdx.x + st];
    __syncthreads();
  }
  if (threadIdx.x == 0) {
    float sum = sm[0];
    float t = scal[0], dt = scal[1];
    float dtc = fminf(dt, 1.0f - t);
    float errn = sqrtf(sum / (float)NELEM);
    bool done = (t >= 1.0f - 1e-7f);
    bool adv = (errn <= 1.0f) && (!done);
    float safe = fmaxf(errn, 1e-10f);
    float factor = fminf(fmaxf(0.9f * powf(safe, -0.2f), 0.2f), 10.0f);
    scal[0] = adv ? (t + dtc) : t;
    scal[1] = done ? dt : (dtc * factor);
    scal[3] = adv ? 1.0f : 0.0f;
    scal[2] = 0.0f;
  }
}

__global__ __launch_bounds__(256) void k_accept(float* __restrict__ y,
                                                const float* __restrict__ y5,
                                                const float* __restrict__ scal) {
  int idx = blockIdx.x * 256 + threadIdx.x;
  if (scal[3] > 0.5f) y[idx] = y5[idx];
}

// ---------------- final linear head: (64,8192) @ wl^T + bl ----------------
__global__ __launch_bounds__(256) void k_head(const float* __restrict__ y,
                                              const float* __restrict__ wl,
                                              const float* __restrict__ bl,
                                              float* __restrict__ out) {
  int gw = (blockIdx.x * 256 + threadIdx.x) >> 5;   // 640 waves
  int lane = threadIdx.x & 31;
  int b = gw / 10, j = gw % 10;
  const float* yb = y + b * 8192;
  const float* wj = wl + j * 8192;
  float s = 0.0f;
  for (int f = lane; f < 8192; f += 32) s += yb[f] * wj[f];
  for (int m = 16; m >= 1; m >>= 1) s += __shfl_xor(s, m, 32);
  if (lane == 0) out[b * 10 + j] = s + bl[j];
}

// ---------------- host launcher ----------------
extern "C" void kernel_launch(void* const* d_in, const int* in_sizes, int n_in,
                              void* d_out, int out_size, void* d_ws, size_t ws_size,
                              hipStream_t stream) {
  const float* x  = (const float*)d_in[0];
  const float* w1 = (const float*)d_in[1];
  const float* b1 = (const float*)d_in[2];
  const float* w2 = (const float*)d_in[3];
  const float* b2 = (const float*)d_in[4];
  const float* w3 = (const float*)d_in[5];
  const float* b3 = (const float*)d_in[6];
  const float* wl = (const float*)d_in[7];
  const float* bl = (const float*)d_in[8];
  float* out = (float*)d_out;

  char* ws = (char*)d_ws;
  size_t off = 0;
  auto alloc = [&](size_t bytes) { char* p = ws + off; off += (bytes + 255) & ~(size_t)255; return p; };
  float*    y    = (float*)alloc(NELEM * 4);
  float*    y5   = (float*)alloc(NELEM * 4);
  float*    kb[7];
  for (int i = 0; i < 7; ++i) kb[i] = (float*)alloc(NELEM * 4);
  _Float16* ys_h = (_Float16*)alloc((size_t)NPOS * 32 * 2);
  _Float16* h1_h = (_Float16*)alloc((size_t)NPOS * 64 * 2);
  _Float16* h2_h = (_Float16*)alloc((size_t)NPOS * 64 * 2);
  _Float16* w1h  = (_Float16*)alloc(2048 * 2);
  _Float16* w2h  = (_Float16*)alloc(36864 * 2);
  _Float16* w3h  = (_Float16*)alloc(1024 * 2);
  float*    w1t  = (float*)alloc(64 * 4);
  float*    tw2  = (float*)alloc(64 * 9 * 4);
  float*    w3t  = (float*)alloc(8 * 4);
  float*    scal = (float*)alloc(4 * 4);
  float*    part = (float*)alloc(2048 * 4);

  hipLaunchKernelGGL(k_init, dim3(2048), dim3(256), 0, stream, y, x, scal);
  hipLaunchKernelGGL(k_prep, dim3(1), dim3(256), 0, stream, w1, w2, w3,
                     w1h, w2h, w3h, w1t, tw2, w3t);

  // dopri5 stage tables: row i = coefficients of k1..k6 for stage input; row 6 = b (y5)
  static const float AC[7][6] = {
    {0, 0, 0, 0, 0, 0},
    {1.0f/5, 0, 0, 0, 0, 0},
    {3.0f/40, 9.0f/40, 0, 0, 0, 0},
    {44.0f/45, -56.0f/15, 32.0f/9, 0, 0, 0},
    {19372.0f/6561, -25360.0f/2187, 64448.0f/6561, -212.0f/729, 0, 0},
    {9017.0f/3168, -355.0f/33, 46732.0f/5247, 49.0f/176, -5103.0f/18656, 0},
    {35.0f/384, 0, 500.0f/1113, 125.0f/192, -2187.0f/6784, 11.0f/84}
  };
  static const float CI[7] = {0.0f, 0.2f, 0.3f, 0.8f, 8.0f/9.0f, 1.0f, 1.0f};

  for (int step = 0; step < 24; ++step) {
    // stages 1..6 -> k1..k6
    for (int s = 0; s < 6; ++s) {
      hipLaunchKernelGGL(k_stage, dim3(8192), dim3(256), 0, stream,
          y, kb[0], kb[1], kb[2], kb[3], kb[4], kb[5],
          AC[s][0], AC[s][1], AC[s][2], AC[s][3], AC[s][4], AC[s][5],
          ys_h, (float*)nullptr, scal);
      hipLaunchKernelGGL(k_conv1, dim3(512), dim3(256), 0, stream,
          ys_h, w1h, b1, w1t, h1_h, scal, CI[s]);
      hipLaunchKernelGGL(k_conv2, dim3(512), dim3(256), 0, stream,
          h1_h, w2h, b2, tw2, h2_h, scal, CI[s]);
      hipLaunchKernelGGL(k_conv3, dim3(512), dim3(256), 0, stream,
          h2_h, w3h, b3, w3t, kb[s], scal, CI[s]);
    }
    // y5 (5th-order solution), then FSAL stage k7 = f(t+dt, y5)
    hipLaunchKernelGGL(k_stage, dim3(8192), dim3(256), 0, stream,
        y, kb[0], kb[1], kb[2], kb[3], kb[4], kb[5],
        AC[6][0], AC[6][1], AC[6][2], AC[6][3], AC[6][4], AC[6][5],
        ys_h, y5, scal);
    hipLaunchKernelGGL(k_conv1, dim3(512), dim3(256), 0, stream,
        ys_h, w1h, b1, w1t, h1_h, scal, CI[6]);
    hipLaunchKernelGGL(k_conv2, dim3(512), dim3(256), 0, stream,
        h1_h, w2h, b2, tw2, h2_h, scal, CI[6]);
    hipLaunchKernelGGL(k_conv3, dim3(512), dim3(256), 0, stream,
        h2_h, w3h, b3, w3t, kb[6], scal, CI[6]);
    // error norm, controller, conditional accept
    hipLaunchKernelGGL(k_err, dim3(2048), dim3(256), 0, stream,
        y, y5, kb[0], kb[2], kb[3], kb[4], kb[5], kb[6], scal, part);
    hipLaunchKernelGGL(k_control, dim3(1), dim3(256), 0, stream, scal, part);
    hipLaunchKernelGGL(k_accept, dim3(2048), dim3(256), 0, stream, y, y5, scal);
  }

  hipLaunchKernelGGL(k_head, dim3(80), dim3(256), 0, stream, y, wl, bl, out);
  (void)in_sizes; (void)n_in; (void)out_size; (void)ws_size;
}
